// Qwen3MoeSparseMoeBlock_50397146251458
// MI455X (gfx1250) — matrix operand order repaired and checked by reading, not compiled, not run on hardware
//
#include <hip/hip_runtime.h>
#include <hip/hip_bf16.h>

// Problem sizes (match reference)
#define S_TOK 512
#define HID   2048
#define NEXP  64
#define IDIM  768
#define TOPK  8

// Tiling
#define TM    32                 // token tile (M)
#define MAXT  (S_TOK / TM)       // max tiles per expert = 16
#define I_PAD 776                // LDS row stride for h tile (bank-conflict-free)

typedef __attribute__((ext_vector_type(16))) __bf16 v16bf;
typedef __attribute__((ext_vector_type(8)))  __bf16 v8bf;
typedef __attribute__((ext_vector_type(8)))  float  v8f;
typedef __attribute__((ext_vector_type(4)))  float  v4f;

// ---- helpers -------------------------------------------------------------

// Load 16 consecutive fp32 and convert to a bf16x16 WMMA B-fragment chunk.
// B layout chosen here (same contiguous k order as A): lane n in [0,16) holds K=0..15 contiguous,
// lane n+16 holds K=16..31 contiguous -> each lane reads 16 contiguous values.
static __device__ __forceinline__ v16bf cvt_f32x16_bf16(const float* __restrict__ p) {
  const v4f* q = (const v4f*)p;
  v4f a = q[0], b = q[1], c = q[2], d = q[3];
  v16bf r;
  r[0]  = (__bf16)a[0]; r[1]  = (__bf16)a[1]; r[2]  = (__bf16)a[2]; r[3]  = (__bf16)a[3];
  r[4]  = (__bf16)b[0]; r[5]  = (__bf16)b[1]; r[6]  = (__bf16)b[2]; r[7]  = (__bf16)b[3];
  r[8]  = (__bf16)c[0]; r[9]  = (__bf16)c[1]; r[10] = (__bf16)c[2]; r[11] = (__bf16)c[3];
  r[12] = (__bf16)d[0]; r[13] = (__bf16)d[1]; r[14] = (__bf16)d[2]; r[15] = (__bf16)d[3];
  return r;
}

// A fragment (16-bit, 16x32): lanes 0-15 hold M=lane, K=k0..k0+15 contiguous;
// lanes 16-31 hold M=lane-16, K=k0+16..k0+31 contiguous.
// Works for both global bf16 rows and LDS rows (generic pointer).
static __device__ __forceinline__ v16bf load_a_frag(const __bf16* row, int k0, int lane) {
  int off = k0 + ((lane & 16) ? 16 : 0);
  v8bf lo = *(const v8bf*)(row + off);
  v8bf hi = *(const v8bf*)(row + off + 8);
  v16bf r;
#pragma unroll
  for (int i = 0; i < 8; ++i) { r[i] = lo[i]; r[8 + i] = hi[i]; }
  return r;
}

static __device__ __forceinline__ v8f wmma_bf16(v16bf a, v16bf b, v8f c) {
  return __builtin_amdgcn_wmma_f32_16x16x32_bf16(false, a, false, b, (short)0, c, false, false);
}

// ---- kernel 1: x -> bf16 copy + zero output ------------------------------

__global__ __launch_bounds__(256) void prep_kernel(const float* __restrict__ x,
                                                   __bf16* __restrict__ xb,
                                                   float* __restrict__ out) {
  int i = blockIdx.x * 256 + threadIdx.x;   // grid sized exactly S*H/256
  xb[i]  = (__bf16)x[i];
  out[i] = 0.0f;
}

// ---- kernel 2: router (logits -> softmax -> top-8 -> renorm) -------------

__global__ __launch_bounds__(256) void router_kernel(const float* __restrict__ x,
                                                     const float* __restrict__ gw,
                                                     int* __restrict__ topk_idx,
                                                     float* __restrict__ topk_w) {
  __shared__ float xs[HID];
  __shared__ float partial[256];
  __shared__ float lg[NEXP];

  const int token = blockIdx.x;
  const int tid   = threadIdx.x;

  // stage x row
  for (int i = tid; i < HID; i += 256) xs[i] = x[(size_t)token * HID + i];
  __syncthreads();

  // 4 chunks x 64 experts
  const int e = tid & 63;
  const int chunk = tid >> 6;
  const float* grow = gw + (size_t)e * HID + chunk * (HID / 4);
  const float* xrow = xs + chunk * (HID / 4);
  float acc = 0.0f;
  for (int k = 0; k < HID / 4; ++k) acc += xrow[k] * grow[k];
  partial[tid] = acc;
  __syncthreads();

  if (tid < NEXP)
    lg[tid] = partial[tid] + partial[tid + 64] + partial[tid + 128] + partial[tid + 192];
  __syncthreads();

  if (tid == 0) {
    // softmax numerator only: renormalized top-k cancels the denominator
    float mx = -1e30f;
    for (int i = 0; i < NEXP; ++i) mx = fmaxf(mx, lg[i]);
    for (int i = 0; i < NEXP; ++i) lg[i] = __expf(lg[i] - mx);

    int   sel_i[TOPK];
    float sel_v[TOPK];
    float ssel = 0.0f;
#pragma unroll
    for (int k = 0; k < TOPK; ++k) {
      float bv = -1.0f; int bi = 0;
      for (int i = 0; i < NEXP; ++i)
        if (lg[i] > bv) { bv = lg[i]; bi = i; }
      sel_i[k] = bi; sel_v[k] = bv; ssel += bv;
      lg[bi] = -1.0f;
    }
    float inv = __builtin_amdgcn_rcpf(ssel);
#pragma unroll
    for (int k = 0; k < TOPK; ++k) {
      topk_idx[token * TOPK + k] = sel_i[k];
      topk_w[token * TOPK + k]   = sel_v[k] * inv;
    }
  }
}

// ---- kernel 3: build per-expert token lists ------------------------------

__global__ __launch_bounds__(256) void build_lists_kernel(const int* __restrict__ topk_idx,
                                                          const float* __restrict__ topk_w,
                                                          int* __restrict__ counts,
                                                          int* __restrict__ offsets,
                                                          int* __restrict__ slot_token,
                                                          float* __restrict__ slot_w) {
  __shared__ int cnt[NEXP];
  __shared__ int off[NEXP];
  __shared__ int cur[NEXP];
  const int tid = threadIdx.x;

  if (tid < NEXP) { cnt[tid] = 0; cur[tid] = 0; }
  __syncthreads();

  for (int i = tid; i < S_TOK * TOPK; i += 256)
    atomicAdd(&cnt[topk_idx[i]], 1);
  __syncthreads();

  if (tid == 0) {
    int acc = 0;
    for (int e2 = 0; e2 < NEXP; ++e2) { off[e2] = acc; acc += cnt[e2]; }
  }
  __syncthreads();

  for (int i = tid; i < S_TOK * TOPK; i += 256) {
    int e   = topk_idx[i];
    int pos = off[e] + atomicAdd(&cur[e], 1);
    slot_token[pos] = i / TOPK;
    slot_w[pos]     = topk_w[i];
  }
  if (tid < NEXP) { counts[tid] = cnt[tid]; offsets[tid] = off[tid]; }
}

// ---- kernel 4: grouped expert SwiGLU FFN (bf16 WMMA) ---------------------

__global__ __launch_bounds__(256) void moe_ffn_kernel(const __bf16* __restrict__ xb,
                                                      const float* __restrict__ w_gate,
                                                      const float* __restrict__ w_up,
                                                      const float* __restrict__ w_down,
                                                      const int* __restrict__ counts,
                                                      const int* __restrict__ offsets,
                                                      const int* __restrict__ slot_token,
                                                      const float* __restrict__ slot_w,
                                                      float* __restrict__ out) {
  __shared__ __align__(16) __bf16 h_lds[TM * I_PAD];   // 32 x 768 h tile (padded)
  __shared__ int   tok_s[TM];
  __shared__ float w_s[TM];

  const int tile = blockIdx.x;     // tile index (x fastest -> same-expert tiles adjacent for L2)
  const int e    = blockIdx.y;     // expert
  const int cnt  = counts[e];
  if (tile * TM >= cnt) return;

  const int tid  = threadIdx.x;
  const int base = offsets[e] + tile * TM;

  if (tid < TM) {
    int s = tile * TM + tid;
    if (s < cnt) { tok_s[tid] = slot_token[base + tid]; w_s[tid] = slot_w[base + tid]; }
    else         { tok_s[tid] = 0;                      w_s[tid] = 0.0f; }
  }
  __syncthreads();

  const int wave = tid >> 5;
  const int lane = tid & 31;
  const int wm   = wave >> 2;        // 0..1  (M half)
  const int wn   = wave & 3;         // 0..3  (N quarter)
  const int nl   = lane & 15;
  const int kb   = (lane & 16) ? 16 : 0;

  const __bf16* xrowA = xb + (size_t)tok_s[wm * 16 + nl] * HID;

  const float* wg_e = w_gate + (size_t)e * IDIM * HID;
  const float* wu_e = w_up   + (size_t)e * IDIM * HID;

  // -------- phase 1: h = silu(x @ wg^T) * (x @ wu^T)  [TM x IDIM] --------
  for (int nc = 0; nc < IDIM; nc += 64) {
    const int n0 = nc + wn * 16;
    v8f accg = {}; v8f accu = {};
    const float* bg_base = wg_e + (size_t)(n0 + nl) * HID + kb;
    const float* bu_base = wu_e + (size_t)(n0 + nl) * HID + kb;
#pragma unroll 4
    for (int k0 = 0; k0 < HID; k0 += 32) {
      v16bf a  = load_a_frag(xrowA, k0, lane);
      v16bf bg = cvt_f32x16_bf16(bg_base + k0);
      accg = wmma_bf16(a, bg, accg);
      v16bf bu = cvt_f32x16_bf16(bu_base + k0);
      accu = wmma_bf16(a, bu, accu);
    }
    // C layout: element r -> (m = wm*16 + r + 8*(lane>=16), n = n0 + nl)
#pragma unroll
    for (int r = 0; r < 8; ++r) {
      float g  = accg[r];
      float u  = accu[r];
      // silu(g)*u with fast v_rcp_f32 (bf16 rounding downstream dwarfs rcp error)
      float hv = g * u * __builtin_amdgcn_rcpf(1.0f + __expf(-g));
      int m = wm * 16 + r + ((lane & 16) ? 8 : 0);
      h_lds[m * I_PAD + (n0 + nl)] = (__bf16)hv;
    }
  }
  __syncthreads();

  // -------- phase 2: out[token] += w * (h @ wd^T)  [TM x HID] ------------
  const float*  wd_e = w_down + (size_t)e * HID * IDIM;
  const __bf16* hrow = &h_lds[(wm * 16 + nl) * I_PAD];

  for (int hn = wn * 16; hn < HID; hn += 64) {
    v8f acc = {};
    const float* bd_base = wd_e + (size_t)(hn + nl) * IDIM + kb;
#pragma unroll 2
    for (int k0 = 0; k0 < IDIM; k0 += 32) {
      v16bf a  = load_a_frag(hrow, k0, lane);
      v16bf bd = cvt_f32x16_bf16(bd_base + k0);
      acc = wmma_bf16(a, bd, acc);
    }
#pragma unroll
    for (int r = 0; r < 8; ++r) {
      int m = wm * 16 + r + ((lane & 16) ? 8 : 0);
      int n = hn + nl;
      atomicAdd(out + (size_t)tok_s[m] * HID + n, acc[r] * w_s[m]);
    }
  }
}

// ---- host launcher -------------------------------------------------------

extern "C" void kernel_launch(void* const* d_in, const int* in_sizes, int n_in,
                              void* d_out, int out_size, void* d_ws, size_t ws_size,
                              hipStream_t stream) {
  (void)in_sizes; (void)n_in; (void)out_size; (void)ws_size;

  const float* x       = (const float*)d_in[0];   // [S, H]
  const float* gw      = (const float*)d_in[1];   // [E, H]
  const float* w_gate  = (const float*)d_in[2];   // [E, I, H]
  const float* w_up    = (const float*)d_in[3];   // [E, I, H]
  const float* w_down  = (const float*)d_in[4];   // [E, H, I]
  float*       out     = (float*)d_out;           // [S, H]

  // workspace carve-out (~2.2 MB total)
  char* ws = (char*)d_ws;
  auto align16 = [](size_t v) { return (v + 15) & ~(size_t)15; };

  __bf16* xb        = (__bf16*)ws;                 ws += align16((size_t)S_TOK * HID * sizeof(__bf16));
  int*    topk_idx  = (int*)ws;                    ws += align16((size_t)S_TOK * TOPK * sizeof(int));
  float*  topk_w    = (float*)ws;                  ws += align16((size_t)S_TOK * TOPK * sizeof(float));
  int*    counts    = (int*)ws;                    ws += align16((size_t)NEXP * sizeof(int));
  int*    offsets   = (int*)ws;                    ws += align16((size_t)NEXP * sizeof(int));
  int*    slot_tok  = (int*)ws;                    ws += align16((size_t)S_TOK * TOPK * sizeof(int));
  float*  slot_w    = (float*)ws;                  ws += align16((size_t)S_TOK * TOPK * sizeof(float));

  // 1) bf16 copy of x + zero out
  prep_kernel<<<(S_TOK * HID) / 256, 256, 0, stream>>>(x, xb, out);

  // 2) router
  router_kernel<<<S_TOK, 256, 0, stream>>>(x, gw, topk_idx, topk_w);

  // 3) per-expert token lists
  build_lists_kernel<<<1, 256, 0, stream>>>(topk_idx, topk_w, counts, offsets, slot_tok, slot_w);

  // 4) grouped expert FFN
  dim3 grid(MAXT, NEXP);
  moe_ffn_kernel<<<grid, 256, 0, stream>>>(xb, w_gate, w_up, w_down,
                                           counts, offsets, slot_tok, slot_w, out);
}
